// MultiHeadAttention_82068235092221
// MI455X (gfx1250) — compile-verified, matrix operand order checked by reading, not run on hardware
//
#include <hip/hip_runtime.h>

typedef __attribute__((ext_vector_type(16))) _Float16 v16h;
typedef __attribute__((ext_vector_type(8)))  _Float16 v8h;
typedef __attribute__((ext_vector_type(8)))  float    v8f;

#define D_MODEL   1024
#define NUM_HEADS 16
#define HEAD_DIM  64
#define BATCH     4
#define SEQ       2048

#if __has_builtin(__builtin_amdgcn_global_load_async_to_lds_b128)
#define USE_ASYNC_LDS 1
#else
#define USE_ASYNC_LDS 0
#endif

static __device__ __forceinline__ v8f v8f_zero() {
  v8f z;
#pragma unroll
  for (int e = 0; e < 8; ++e) z[e] = 0.0f;
  return z;
}

// Cross-lane exchange lane L <-> lane L^16 (wave32) without LDS traffic.
static __device__ __forceinline__ float xor16(float v) {
#if __has_builtin(__builtin_amdgcn_permlanex16)
  int i = __builtin_bit_cast(int, v);
  i = __builtin_amdgcn_permlanex16(i, i, 0x76543210, 0xfedcba98, false, false);
  return __builtin_bit_cast(float, i);
#else
  return __shfl_xor(v, 16, 32);
#endif
}

// Async bulk copy 16B/lane global -> LDS (GLOBAL_LOAD_ASYNC_TO_LDS_B128,
// tracked by ASYNCcnt). LDS operand = low 32 bits of the generic pointer,
// which per the flat-aperture rules is the LDS byte offset.
static __device__ __forceinline__ void async_copy_b128(const _Float16* g, _Float16* l) {
#if USE_ASYNC_LDS
  unsigned lo = (unsigned)(unsigned long long)(uintptr_t)l;
  asm volatile("global_load_async_to_lds_b128 %0, %1, off"
               :: "v"(lo), "v"(g) : "memory");
#else
  *(v8h*)l = *(const v8h*)g;
#endif
}

static __device__ __forceinline__ void async_fence() {
#if USE_ASYNC_LDS
  asm volatile("s_wait_asynccnt 0" ::: "memory");
#endif
}

// WMMA A/B fragment from a row-major tile: element e of lane L holds
// M[row][kbase + (e>>3)*16 + 8*(L>>4) + (e&7)]  -> two 16B loads per lane.
static __device__ __forceinline__ v16h frag_ld_row(const _Float16* base, int row,
                                                   int ld, int kbase, int lane) {
  const int hi8 = (lane >> 4) << 3;
  const _Float16* p = base + row * ld + kbase + hi8;
  v8h lo = *(const v8h*)(p);
  v8h hi = *(const v8h*)(p + 16);
  v16h r;
#pragma unroll
  for (int i = 0; i < 8; ++i) { r[i] = lo[i]; r[i + 8] = hi[i]; }
  return r;
}

// A-fragment of the TRANSPOSE of a row-major tile (for V^T): element e of lane
// L holds M[krow0 + (e>>3)*16 + 8*(L>>4) + (e&7)][col].
static __device__ __forceinline__ v16h frag_ld_col(const _Float16* base, int col,
                                                   int ld, int krow0, int lane) {
  const int hi8 = (lane >> 4) << 3;
  v16h r;
#pragma unroll
  for (int i = 0; i < 8; ++i) {
    r[i]     = base[(krow0 + hi8 + i) * ld + col];
    r[i + 8] = base[(krow0 + 16 + hi8 + i) * ld + col];
  }
  return r;
}

// C = A[M,K] @ W[N,K]^T + bias. Block = 128 thr = 4 waves; block tile 128x64,
// wave tile 32x64 (8 C-fragments) so each B fragment feeds two WMMAs. K-step 32.
template <typename TA, bool OUT_HALF>
__global__ __launch_bounds__(128)
void gemm_xwt_bias(const TA* __restrict__ A, const float* __restrict__ W,
                   const float* __restrict__ bias, void* __restrict__ outp,
                   int M, int N, int K) {
  (void)M;
  __shared__ _Float16 Alds[128 * 40];
  __shared__ _Float16 Wlds[64 * 40];
  const int lane = threadIdx.x & 31;
  const int wave = threadIdx.x >> 5;
  const int l15  = lane & 15;
  const int hi   = lane >> 4;
  const int m0   = blockIdx.y * 128;
  const int n0   = blockIdx.x * 64;

  v8f acc[2][4];
#pragma unroll
  for (int mi = 0; mi < 2; ++mi)
#pragma unroll
    for (int j = 0; j < 4; ++j) acc[mi][j] = v8f_zero();

  const int ar = threadIdx.x;             // A staged row 0..127 (32 elems/row)
  const int wr = threadIdx.x >> 1;        // W staged row 0..63
  const int wc = (threadIdx.x & 1) * 16;  // W staged col base 0/16

  for (int kk = 0; kk < K; kk += 32) {
    const TA*    ap = A + (size_t)(m0 + ar) * K + kk;
    const float* wp = W + (size_t)(n0 + wr) * K + kk + wc;
#pragma unroll
    for (int i = 0; i < 32; ++i) Alds[ar * 40 + i] = (_Float16)ap[i];
#pragma unroll
    for (int i = 0; i < 16; ++i) Wlds[wr * 40 + wc + i] = (_Float16)wp[i];
    __syncthreads();

    const v16h a0 = frag_ld_row(Alds, wave * 32 + l15,      40, 0, lane);
    const v16h a1 = frag_ld_row(Alds, wave * 32 + 16 + l15, 40, 0, lane);
#pragma unroll
    for (int j = 0; j < 4; ++j) {
      const v16h bf = frag_ld_row(Wlds, j * 16 + l15, 40, 0, lane);  // B[k][n]=W[n][k]
      acc[0][j] = __builtin_amdgcn_wmma_f32_16x16x32_f16(
          false, a0, false, bf, (short)0, acc[0][j], false, false);
      acc[1][j] = __builtin_amdgcn_wmma_f32_16x16x32_f16(
          false, a1, false, bf, (short)0, acc[1][j], false, false);
    }
    __syncthreads();
  }

#pragma unroll
  for (int mi = 0; mi < 2; ++mi) {
#pragma unroll
    for (int j = 0; j < 4; ++j) {
      const int n = n0 + j * 16 + l15;
      const float bj = bias[n];
      const int mb = m0 + wave * 32 + mi * 16 + hi * 8;
      if (OUT_HALF) {
        _Float16* out = (_Float16*)outp;
#pragma unroll
        for (int e = 0; e < 8; ++e)
          out[(size_t)(mb + e) * N + n] = (_Float16)(acc[mi][j][e] + bj);
      } else {
        float* out = (float*)outp;
#pragma unroll
        for (int e = 0; e < 8; ++e)
          out[(size_t)(mb + e) * N + n] = acc[mi][j][e] + bj;
      }
    }
  }
}

// Flash attention. Block = 4 waves; each wave owns 16 queries of one (b,h).
// 64-key KV tiles staged to LDS with async b128 copies, shared by the 4 waves.
// Computes S^T = K*Q^T (per-query stats need one permlanex16); the S^T C/D
// layout packs per-lane into the B-operand fragments of P^T for ctx^T = V^T*P^T.
__global__ __launch_bounds__(128)
void flash_attn(const _Float16* __restrict__ Q, const _Float16* __restrict__ Kp,
                const _Float16* __restrict__ V, _Float16* __restrict__ CTX) {
  __shared__ _Float16 Klds[64 * 72];
  __shared__ _Float16 Vlds[64 * 72];
  const int lane = threadIdx.x & 31;
  const int wave = threadIdx.x >> 5;
  const int l15  = lane & 15;
  const int hi   = lane >> 4;
  const int b    = blockIdx.y >> 4;
  const int h    = blockIdx.y & 15;
  const int q0   = blockIdx.x * 64 + wave * 16;
  const size_t base = (size_t)b * SEQ * D_MODEL + (size_t)h * HEAD_DIM;

  // Persistent Q fragments (double as B operand for Q^T).
  const _Float16* qp = Q + base + (size_t)q0 * D_MODEL;
  const v16h qf0 = frag_ld_row(qp, l15, D_MODEL, 0, lane);
  const v16h qf1 = frag_ld_row(qp, l15, D_MODEL, 32, lane);

  v8f ctx[4];  // ctx^T: d-rows 16j..16j+15, q-col = l15
#pragma unroll
  for (int j = 0; j < 4; ++j) ctx[j] = v8f_zero();
  float m_run = -1e30f, l_run = 0.0f;
  const float scale = 0.125f;  // 1/sqrt(HEAD_DIM)

  for (int kv = 0; kv < SEQ; kv += 64) {
    // Cooperative async staging: 64 keys x 64 halves for K and V.
#pragma unroll
    for (int c = threadIdx.x; c < 512; c += 128) {
      const int key = c >> 3;
      const int off = (c & 7) * 8;
      const size_t g = base + (size_t)(kv + key) * D_MODEL + off;
      async_copy_b128(&Kp[g], &Klds[key * 72 + off]);
      async_copy_b128(&V[g],  &Vlds[key * 72 + off]);
    }
    async_fence();
    __syncthreads();

    // S^T[key][q]: 4 key sub-tiles x chained Dh=64.
    v8f sS[4];
#pragma unroll
    for (int s = 0; s < 4; ++s) {
      const v16h ka = frag_ld_row(Klds, s * 16 + l15, 72, 0, lane);
      const v16h kb = frag_ld_row(Klds, s * 16 + l15, 72, 32, lane);
      sS[s] = __builtin_amdgcn_wmma_f32_16x16x32_f16(
          false, ka, false, qf0, (short)0, v8f_zero(), false, false);
      sS[s] = __builtin_amdgcn_wmma_f32_16x16x32_f16(
          false, kb, false, qf1, (short)0, sS[s], false, false);
    }

    // Online softmax in fp32. Lane pair (L, L^16) holds one query column.
    float mt = -1e30f;
#pragma unroll
    for (int s = 0; s < 4; ++s)
#pragma unroll
      for (int e = 0; e < 8; ++e) {
        sS[s][e] *= scale;
        mt = fmaxf(mt, sS[s][e]);
      }
    mt = fmaxf(mt, xor16(mt));
    const float m_new = fmaxf(m_run, mt);
    const float corr  = __expf(m_run - m_new);
    float lt = 0.0f;
    v16h pf0, pf1;  // B-operand fragments of P^T — per-lane pack, no shuffles
#pragma unroll
    for (int e = 0; e < 8; ++e) {
      const float p0 = __expf(sS[0][e] - m_new);
      const float p1 = __expf(sS[1][e] - m_new);
      const float p2 = __expf(sS[2][e] - m_new);
      const float p3 = __expf(sS[3][e] - m_new);
      lt += (p0 + p1) + (p2 + p3);
      pf0[e] = (_Float16)p0; pf0[e + 8] = (_Float16)p1;
      pf1[e] = (_Float16)p2; pf1[e + 8] = (_Float16)p3;
    }
    lt += xor16(lt);
    l_run = l_run * corr + lt;
    m_run = m_new;

#pragma unroll
    for (int j = 0; j < 4; ++j)
#pragma unroll
      for (int e = 0; e < 8; ++e) ctx[j][e] *= corr;

    // ctx^T += V^T * P^T over 64 keys (2 chained WMMAs per 16-d sub-tile).
#pragma unroll
    for (int j = 0; j < 4; ++j) {
      const v16h vf0 = frag_ld_col(Vlds, j * 16 + l15, 72, 0,  lane);
      ctx[j] = __builtin_amdgcn_wmma_f32_16x16x32_f16(
          false, vf0, false, pf0, (short)0, ctx[j], false, false);
      const v16h vf1 = frag_ld_col(Vlds, j * 16 + l15, 72, 32, lane);
      ctx[j] = __builtin_amdgcn_wmma_f32_16x16x32_f16(
          false, vf1, false, pf1, (short)0, ctx[j], false, false);
    }
    __syncthreads();
  }

  // Normalize and store ctx in [B, S, H*Dh] f16 layout; 16B stores per fragment.
  const float inv = 1.0f / l_run;
  _Float16* cp = CTX + base + (size_t)(q0 + l15) * D_MODEL;
#pragma unroll
  for (int j = 0; j < 4; ++j) {
    v8h o;
#pragma unroll
    for (int e = 0; e < 8; ++e) o[e] = (_Float16)(ctx[j][e] * inv);
    *(v8h*)&cp[j * 16 + hi * 8] = o;
  }
}

extern "C" void kernel_launch(void* const* d_in, const int* in_sizes, int n_in,
                              void* d_out, int out_size, void* d_ws, size_t ws_size,
                              hipStream_t stream) {
  (void)in_sizes; (void)n_in; (void)out_size; (void)ws_size;
  const float* query = (const float*)d_in[0];
  const float* key   = (const float*)d_in[1];
  const float* value = (const float*)d_in[2];
  const float* Wq = (const float*)d_in[3];
  const float* bq = (const float*)d_in[4];
  const float* Wk = (const float*)d_in[5];
  const float* bk = (const float*)d_in[6];
  const float* Wv = (const float*)d_in[7];
  const float* bv = (const float*)d_in[8];
  const float* Wo = (const float*)d_in[9];
  const float* bo = (const float*)d_in[10];
  float* out = (float*)d_out;

  const size_t tok = (size_t)BATCH * SEQ;  // 8192
  _Float16* q16 = (_Float16*)d_ws;
  _Float16* k16 = q16 + tok * D_MODEL;
  _Float16* v16 = k16 + tok * D_MODEL;
  _Float16* c16 = v16 + tok * D_MODEL;     // total 64 MB of d_ws

  dim3 blk(128);
  dim3 ggrid(D_MODEL / 64, (int)tok / 128);  // (16, 64)
  gemm_xwt_bias<float, true><<<ggrid, blk, 0, stream>>>(query, Wq, bq, q16, (int)tok, D_MODEL, D_MODEL);
  gemm_xwt_bias<float, true><<<ggrid, blk, 0, stream>>>(key,   Wk, bk, k16, (int)tok, D_MODEL, D_MODEL);
  gemm_xwt_bias<float, true><<<ggrid, blk, 0, stream>>>(value, Wv, bv, v16, (int)tok, D_MODEL, D_MODEL);

  dim3 agrid(SEQ / 64, BATCH * NUM_HEADS);  // (32, 64)
  flash_attn<<<agrid, blk, 0, stream>>>(q16, k16, v16, c16);

  gemm_xwt_bias<_Float16, false><<<ggrid, blk, 0, stream>>>(c16, Wo, bo, out, (int)tok, D_MODEL, D_MODEL);
}
